// Representation_12326556140090
// MI455X (gfx1250) — matrix operand order repaired and checked by reading, not compiled, not run on hardware
//
#include <hip/hip_runtime.h>
#include <hip/hip_bf16.h>

// ---------------------------------------------------------------------------
// GraphSAGE residual GNN forward on MI455X (gfx1250), wave32 + WMMA bf16.
//
// Bandwidth-bound (~106 GFLOP of 128-wide GEMMs, ~30 passes over a 25.6MB
// bf16 activation tensor, everything L2-resident in 192MB L2):
//   * activations bf16, f32 WMMA accumulation (v_wmma_f32_16x16x32_bf16)
//   * all weights converted ONCE per launch into pre-transposed bf16 buffers
//     whose layout matches the LDS pitch -> GEMM staging is a pure linear
//     copy done with GLOBAL_LOAD_ASYNC_TO_LDS_B128 (+ s_wait_asynccnt)
//   * fused SAGE GEMM: double-buffered LDS holds Ws and Wn behind ONE barrier
//   * bias / ELU / residual-add / inv-degree scaling fused into GEMM
// ---------------------------------------------------------------------------

#define GNN_N 100000
#define GNN_E 400000
#define IN_F  53
#define H_F   128
#define OUT_F 256

typedef __attribute__((ext_vector_type(16))) __bf16 v16bf;
typedef __attribute__((ext_vector_type(8)))  float  v8f;

union BFrag {
    v16bf          v;
    unsigned int   d[8];
    unsigned short u[16];
};

#define LDSP_DW   68            // dword pitch of one weight row in LDS
#define LDSP_H    (2 * LDSP_DW) // half pitch (136)
#define WCHUNK_H  (H_F * LDSP_H)       // halves per converted weight (17408)
#define WCHUNK_B  (WCHUNK_H * 2)       // bytes per converted weight (34816)
#define WCOPY_X16 (WCHUNK_B / 16)      // 16B packets per weight copy (2176)

__device__ __forceinline__ unsigned short f2bf(float f) {
    unsigned u = __float_as_uint(f);
    unsigned r = u + 0x7FFFu + ((u >> 16) & 1u);   // round-to-nearest-even
    return (unsigned short)(r >> 16);
}
__device__ __forceinline__ float bf2f(unsigned short h) {
    return __uint_as_float(((unsigned)h) << 16);
}
__device__ __forceinline__ float elu1(float x) {
    return x > 0.0f ? x : (__expf(x) - 1.0f);
}

// Issue an async global->LDS copy of one converted weight (34816 B, linear).
// gfx1250: GLOBAL_LOAD_ASYNC_TO_LDS_B128, tracked by ASYNCcnt.
__device__ __forceinline__ void async_copy_w(const unsigned short* gsrc,
                                             unsigned ldsoff, int tid) {
    unsigned long long gbase = (unsigned long long)gsrc;
    for (int i = tid; i < WCOPY_X16; i += 256) {
        unsigned      off  = (unsigned)i * 16u;
        unsigned      ldst = ldsoff + off;
        unsigned long long ga = gbase + (unsigned long long)off;
        asm volatile("global_load_async_to_lds_b128 %0, %1, off"
                     :: "v"(ldst), "v"(ga) : "memory");
    }
}
__device__ __forceinline__ void async_wait0() {
    asm volatile("s_wait_asynccnt 0x0" ::: "memory");
}

// ---------------------------------------------------------------------------
// One-time weight preparation: f32 (k,n) -> bf16 transposed (n,k) with
// LDS-matching pitch of 136 halves. 32 chunks: 28 layer mats + 4 post chunks.
// ---------------------------------------------------------------------------
struct WChunksArg {
    const float* p[32];
    int          stride[32];
};

__global__ void __launch_bounds__(256)
convert_weights_kernel(WChunksArg wa, unsigned short* __restrict__ dstbase) {
    int c = blockIdx.x;
    const float* S = wa.p[c];
    int stride     = wa.stride[c];
    unsigned short* D = dstbase + (size_t)c * WCHUNK_H;
    for (int idx = threadIdx.x; idx < H_F * H_F; idx += 256) {
        int k = idx >> 7, n = idx & 127;
        D[n * LDSP_H + k] = f2bf(S[k * stride + n]);
    }
}

// ---------------------------------------------------------------------------
// Degree / scatter kernels (L2-resident f32 atomics)
// ---------------------------------------------------------------------------
__global__ void deg_count_kernel(const int* __restrict__ dst,
                                 float* __restrict__ deg, int nedges) {
    int e = blockIdx.x * blockDim.x + threadIdx.x;
    if (e < nedges) atomicAdd(&deg[dst[e]], 1.0f);
}

__global__ void inv_deg_kernel(const float* __restrict__ deg,
                               float* __restrict__ invdeg, int n) {
    int i = blockIdx.x * blockDim.x + threadIdx.x;
    if (i < n) invdeg[i] = 1.0f / fmaxf(deg[i], 1.0f);
}

__global__ void scatter_add_kernel(const unsigned short* __restrict__ h,
                                   const int* __restrict__ src,
                                   const int* __restrict__ dst,
                                   float* __restrict__ agg, int nedges) {
    int t = blockIdx.x * blockDim.x + threadIdx.x;
    int e = t >> 5;
    if (e >= nedges) return;
    int c = (t & 31) * 4;
    int s = src[e], d = dst[e];
    const unsigned short* hp = h + (size_t)s * H_F + c;
    float* ap = agg + (size_t)d * H_F + c;
    atomicAdd(ap + 0, bf2f(hp[0]));
    atomicAdd(ap + 1, bf2f(hp[1]));
    atomicAdd(ap + 2, bf2f(hp[2]));
    atomicAdd(ap + 3, bf2f(hp[3]));
}

// ---------------------------------------------------------------------------
// Pre layer: h = elu(X @ Wpre + b); h_skip = elu(h).  K=53 -> plain VALU.
// ---------------------------------------------------------------------------
__global__ void __launch_bounds__(128)
pre_elu_kernel(const float* __restrict__ X, const float* __restrict__ W,
               const float* __restrict__ b, unsigned short* __restrict__ h,
               unsigned short* __restrict__ hskip, int n) {
    __shared__ float xs[56];
    int i = blockIdx.x;
    int j = threadIdx.x;
    if (i >= n) return;
    if (j < IN_F) xs[j] = X[(size_t)i * IN_F + j];
    __syncthreads();
    float acc = b[j];
    #pragma unroll 1
    for (int k = 0; k < IN_F; ++k) acc = fmaf(xs[k], W[k * H_F + j], acc);
    float e1 = elu1(acc);
    h[(size_t)i * H_F + j]     = f2bf(e1);
    hskip[(size_t)i * H_F + j] = f2bf(elu1(e1));
}

// ---------------------------------------------------------------------------
// Fused 128-wide WMMA GEMM:
//   out = act( A@Wt1 [+ (A2*invdeg)@Wt2] [+ bias] [+ addend] )
// Wt*: pre-converted bf16 transposed (pitch 136 halves) -> async LDS staging.
// Block = 256 thr = 8 waves; wave owns 16 rows x 128 cols.
// ---------------------------------------------------------------------------
__global__ void __launch_bounds__(256)
gemm128_wmma(const unsigned short* __restrict__ A,
             const float* __restrict__ A2,
             const float* __restrict__ invdeg,
             const unsigned short* __restrict__ Wt1,
             const unsigned short* __restrict__ Wt2,
             const float* __restrict__ bias,
             const unsigned short* __restrict__ addend,
             unsigned short* __restrict__ out,
             int nrows, int act) {
    __shared__ __align__(16) unsigned int ldsW[2 * H_F * LDSP_DW]; // 2x34816B

    const int tid  = threadIdx.x;
    const int lane = tid & 31;
    const int wave = tid >> 5;
    const int half = lane >> 4;
    const int lmod = lane & 15;

    // ---- async-stage weights into LDS (both at once for the SAGE case) ----
    const unsigned ldsbase = (unsigned)(uintptr_t)(&ldsW[0]);
    async_copy_w(Wt1, ldsbase, tid);
    if (Wt2) async_copy_w(Wt2, ldsbase + (unsigned)WCHUNK_B, tid);
    async_wait0();
    __syncthreads();

    const int row0 = (blockIdx.x * 8 + wave) * 16;
    int m = row0 + lmod;
    if (m >= nrows) m = nrows - 1;              // clamp (stores are guarded)

    // ---- A fragments: 4 K-steps of 16x32 bf16, contiguous k = 16*half + s ----
    const unsigned int* arow = (const unsigned int*)A + (size_t)m * (H_F / 2);
    BFrag afr[4];
    #pragma unroll
    for (int kt = 0; kt < 4; ++kt) {
        #pragma unroll
        for (int v = 0; v < 8; ++v) {
            int kd = v + half * 8 + kt * 16;
            afr[kt].d[v] = arow[kd];
        }
    }

    v8f acc[8] = {};

    // ---- dense pass: acc += A @ Wt1 ----
    #pragma unroll
    for (int nt = 0; nt < 8; ++nt) {
        const unsigned int* bp = ldsW + (nt * 16 + lmod) * LDSP_DW + half * 8;
        #pragma unroll
        for (int kt = 0; kt < 4; ++kt) {
            BFrag bfr;
            #pragma unroll
            for (int j = 0; j < 8; ++j) bfr.d[j] = bp[kt * 16 + j];
            acc[nt] = __builtin_amdgcn_wmma_f32_16x16x32_bf16(
                false, afr[kt].v, false, bfr.v, (short)0, acc[nt],
                false, false);
        }
    }

    // ---- neighbor pass: acc += (A2 * invdeg) @ Wt2 (already staged) ----
    if (Wt2) {
        float scale = invdeg ? invdeg[m] : 1.0f;
        const float* a2row = A2 + (size_t)m * H_F;
        BFrag a2fr[4];
        #pragma unroll
        for (int kt = 0; kt < 4; ++kt) {
            #pragma unroll
            for (int v = 0; v < 8; ++v) {
                int kd = v + half * 8 + kt * 16;
                a2fr[kt].u[2 * v]     = f2bf(a2row[2 * kd]     * scale);
                a2fr[kt].u[2 * v + 1] = f2bf(a2row[2 * kd + 1] * scale);
            }
        }
        #pragma unroll
        for (int nt = 0; nt < 8; ++nt) {
            const unsigned int* bp = ldsW + H_F * LDSP_DW +
                                     (nt * 16 + lmod) * LDSP_DW + half * 8;
            #pragma unroll
            for (int kt = 0; kt < 4; ++kt) {
                BFrag bfr;
                #pragma unroll
                for (int j = 0; j < 8; ++j) bfr.d[j] = bp[kt * 16 + j];
                acc[nt] = __builtin_amdgcn_wmma_f32_16x16x32_bf16(
                    false, a2fr[kt].v, false, bfr.v, (short)0, acc[nt],
                    false, false);
            }
        }
    }

    // ---- epilogue: bias + residual + ELU, store bf16 ----
    #pragma unroll
    for (int nt = 0; nt < 8; ++nt) {
        int n = nt * 16 + lmod;
        float bv = bias ? bias[n] : 0.0f;
        #pragma unroll
        for (int v = 0; v < 8; ++v) {
            int mm = row0 + v + 8 * half;
            if (mm < nrows) {
                float c = acc[nt][v] + bv;
                if (addend) c += bf2f(addend[(size_t)mm * H_F + n]);
                if (act) c = elu1(c);
                out[(size_t)mm * H_F + n] = f2bf(c);
            }
        }
    }
}

// ---------------------------------------------------------------------------
// Post layer: out[N,256] = concat(h, h_skip)[N,256] @ Wpost[256,256] + b.
// WtPost = 4 pre-converted chunks, chunk (ny,kk) = WtPost + (ny*2+kk)*WCHUNK_H.
// grid.y selects 128-wide column half; kk walks h then h_skip K-halves.
// ---------------------------------------------------------------------------
__global__ void __launch_bounds__(256)
gemm_post_wmma(const unsigned short* __restrict__ H,
               const unsigned short* __restrict__ Hskip,
               const unsigned short* __restrict__ WtPost,
               const float* __restrict__ bias,
               float* __restrict__ out,
               int nrows) {
    __shared__ __align__(16) unsigned int ldsW[H_F * LDSP_DW];

    const int tid  = threadIdx.x;
    const int lane = tid & 31;
    const int wave = tid >> 5;
    const int half = lane >> 4;
    const int lmod = lane & 15;
    const int ny   = blockIdx.y;                 // column half

    const int row0 = (blockIdx.x * 8 + wave) * 16;
    int m = row0 + lmod;
    if (m >= nrows) m = nrows - 1;

    const unsigned ldsbase = (unsigned)(uintptr_t)(&ldsW[0]);

    v8f acc[8] = {};

    for (int kk = 0; kk < 2; ++kk) {
        __syncthreads();                         // previous LDS reads done
        async_copy_w(WtPost + (size_t)(ny * 2 + kk) * WCHUNK_H, ldsbase, tid);
        async_wait0();
        __syncthreads();

        const unsigned short* Asrc = kk ? Hskip : H;
        const unsigned int* arow =
            (const unsigned int*)Asrc + (size_t)m * (H_F / 2);
        BFrag afr[4];
        #pragma unroll
        for (int kt = 0; kt < 4; ++kt) {
            #pragma unroll
            for (int v = 0; v < 8; ++v) {
                int kd = v + half * 8 + kt * 16;
                afr[kt].d[v] = arow[kd];
            }
        }
        #pragma unroll
        for (int nt = 0; nt < 8; ++nt) {
            const unsigned int* bp =
                ldsW + (nt * 16 + lmod) * LDSP_DW + half * 8;
            #pragma unroll
            for (int kt = 0; kt < 4; ++kt) {
                BFrag bfr;
                #pragma unroll
                for (int j = 0; j < 8; ++j) bfr.d[j] = bp[kt * 16 + j];
                acc[nt] = __builtin_amdgcn_wmma_f32_16x16x32_bf16(
                    false, afr[kt].v, false, bfr.v, (short)0, acc[nt],
                    false, false);
            }
        }
    }

    #pragma unroll
    for (int nt = 0; nt < 8; ++nt) {
        int n = ny * 128 + nt * 16 + lmod;
        float bv = bias[n];
        #pragma unroll
        for (int v = 0; v < 8; ++v) {
            int mm = row0 + v + 8 * half;
            if (mm < nrows)
                out[(size_t)mm * OUT_F + n] = acc[nt][v] + bv;
        }
    }
}

// ---------------------------------------------------------------------------
// Host orchestration
// ---------------------------------------------------------------------------
extern "C" void kernel_launch(void* const* d_in, const int* in_sizes, int n_in,
                              void* d_out, int out_size, void* d_ws, size_t ws_size,
                              hipStream_t stream) {
    const int N = GNN_N, E = GNN_E;

    // ---- input mapping: params flattened in JAX pytree (alphabetical) order;
    //      tolerate either top-level ordering via size check -----------------
    const float* X;
    const int *src, *dst;
    int base;
    if (in_sizes[0] == E) {          // alphabetical: dst, in_feature, params..., src
        dst  = (const int*)d_in[0];
        X    = (const float*)d_in[1];
        base = 2;
        src  = (const int*)d_in[n_in - 1];
    } else {                          // insertion: in_feature, params..., src, dst
        X    = (const float*)d_in[0];
        base = 1;
        src  = (const int*)d_in[n_in - 2];
        dst  = (const int*)d_in[n_in - 1];
    }
    // 52 param leaves: 3 blocks x [sage1(Wn,Ws,b) sage2(Wn,Ws,b) si1(W,b)
    // si2(W,b) skip1(W,b) skip2(W,b)] ; convs x2 (Wn,Ws,b) ; post(W,b) ; pre(W,b)
    const float* q[52];
    for (int i = 0; i < 52; ++i) q[i] = (const float*)d_in[base + i];
    const float* preW  = q[50];
    const float* preB  = q[51];
    const float* postW = q[48];
    const float* postB = q[49];

    // ---- workspace layout ----
    char*  ws = (char*)d_ws;
    size_t o  = 0;
    auto alloc = [&](size_t bytes) {
        void* r = ws + o;
        o = (o + bytes + 255) & ~(size_t)255;
        return r;
    };
    float* deg    = (float*)alloc((size_t)N * 4);
    float* invdeg = (float*)alloc((size_t)N * 4);
    float* agg    = (float*)alloc((size_t)N * H_F * 4);
    unsigned short* wbf = (unsigned short*)alloc((size_t)32 * WCHUNK_B);
    unsigned short* h0    = (unsigned short*)alloc((size_t)N * H_F * 2);
    unsigned short* h1    = (unsigned short*)alloc((size_t)N * H_F * 2);
    unsigned short* hs    = (unsigned short*)alloc((size_t)N * H_F * 2);
    unsigned short* t1    = (unsigned short*)alloc((size_t)N * H_F * 2);
    unsigned short* t2    = (unsigned short*)alloc((size_t)N * H_F * 2);
    unsigned short* hskip = (unsigned short*)alloc((size_t)N * H_F * 2);
    (void)ws_size; (void)out_size;

    auto wt = [&](int idx) -> const unsigned short* {
        return wbf + (size_t)idx * WCHUNK_H;
    };

    // ---- one-time weight conversion (bf16, transposed, LDS pitch) ----
    // chunk order: per block b: Wn1 Ws1 Wn2 Ws2 si1 si2 skip1 skip2 ;
    //              convs: Wn Ws x2 ; post: (ny,kk) = 28 + ny*2 + kk
    WChunksArg wa;
    int ci = 0;
    for (int b = 0; b < 3; ++b) {
        const float* const* Q = q + b * 14;
        const float* mats[8] = {Q[0], Q[1], Q[3], Q[4], Q[6], Q[8], Q[10], Q[12]};
        for (int j = 0; j < 8; ++j) { wa.p[ci] = mats[j]; wa.stride[ci] = H_F; ++ci; }
    }
    for (int c = 0; c < 2; ++c) {
        wa.p[ci] = q[42 + c * 3]; wa.stride[ci] = H_F; ++ci;   // Wn
        wa.p[ci] = q[43 + c * 3]; wa.stride[ci] = H_F; ++ci;   // Ws
    }
    for (int ny = 0; ny < 2; ++ny)
        for (int kk = 0; kk < 2; ++kk) {
            wa.p[ci] = postW + (size_t)kk * 128 * OUT_F + ny * 128;
            wa.stride[ci] = OUT_F; ++ci;
        }
    convert_weights_kernel<<<32, 256, 0, stream>>>(wa, wbf);

    const int GB = (N + 127) / 128;                      // gemm blocks
    const int SB = (int)(((size_t)E * 32 + 255) / 256);  // scatter blocks

    // ---- degree + pre layer ----
    hipMemsetAsync(deg, 0, (size_t)N * 4, stream);
    deg_count_kernel<<<(E + 255) / 256, 256, 0, stream>>>(dst, deg, E);
    inv_deg_kernel<<<(N + 255) / 256, 256, 0, stream>>>(deg, invdeg, N);
    pre_elu_kernel<<<N, 128, 0, stream>>>(X, preW, preB, h0, hskip, N);

    unsigned short* h  = h0;
    unsigned short* hn = h1;

    // ---- 3 residual blocks ----
    for (int b = 0; b < 3; ++b) {
        const float* const* Q = q + b * 14;
        const int b8 = b * 8;
        // hs = Linear(ELU(Linear(h)))
        gemm128_wmma<<<GB, 256, 0, stream>>>(h, nullptr, nullptr, wt(b8 + 6),
                                             nullptr, Q[11], nullptr, t1, N, 1);
        gemm128_wmma<<<GB, 256, 0, stream>>>(t1, nullptr, nullptr, wt(b8 + 7),
                                             nullptr, Q[13], nullptr, hs, N, 0);
        // h1 = ELU(sage1(h))
        hipMemsetAsync(agg, 0, (size_t)N * H_F * 4, stream);
        scatter_add_kernel<<<SB, 256, 0, stream>>>(h, src, dst, agg, E);
        gemm128_wmma<<<GB, 256, 0, stream>>>(h, agg, invdeg, wt(b8 + 1) /*Ws*/,
                                             wt(b8 + 0) /*Wn*/, Q[2], nullptr,
                                             t1, N, 1);
        // h1 = ELU(Linear(ELU(Linear(h1))))
        gemm128_wmma<<<GB, 256, 0, stream>>>(t1, nullptr, nullptr, wt(b8 + 4),
                                             nullptr, Q[7], nullptr, t2, N, 1);
        gemm128_wmma<<<GB, 256, 0, stream>>>(t2, nullptr, nullptr, wt(b8 + 5),
                                             nullptr, Q[9], nullptr, t1, N, 1);
        // h = ELU(hs + sage2(h1))
        hipMemsetAsync(agg, 0, (size_t)N * H_F * 4, stream);
        scatter_add_kernel<<<SB, 256, 0, stream>>>(t1, src, dst, agg, E);
        gemm128_wmma<<<GB, 256, 0, stream>>>(t1, agg, invdeg, wt(b8 + 3) /*Ws*/,
                                             wt(b8 + 2) /*Wn*/, Q[5], hs, hn, N, 1);
        unsigned short* tmp = h; h = hn; hn = tmp;
    }

    // ---- 2 conv layers: h = ELU(sage(h)) ----
    for (int c = 0; c < 2; ++c) {
        const float* bb = q[44 + c * 3];
        hipMemsetAsync(agg, 0, (size_t)N * H_F * 4, stream);
        scatter_add_kernel<<<SB, 256, 0, stream>>>(h, src, dst, agg, E);
        gemm128_wmma<<<GB, 256, 0, stream>>>(h, agg, invdeg,
                                             wt(24 + c * 2 + 1) /*Ws*/,
                                             wt(24 + c * 2) /*Wn*/, bb,
                                             nullptr, hn, N, 1);
        unsigned short* tmp = h; h = hn; hn = tmp;
    }

    // ---- post: concat(h, h_skip) @ Wpost + b -> f32 out ----
    gemm_post_wmma<<<dim3(GB, 2), 256, 0, stream>>>(h, hskip, wt(28), postB,
                                                    (float*)d_out, N);
}
